// SelfAttention_11527692222727
// MI455X (gfx1250) — compile-verified, matrix operand order checked by reading, not run on hardware
//
#include <hip/hip_runtime.h>

typedef __attribute__((ext_vector_type(16))) _Float16 v16h;
typedef __attribute__((ext_vector_type(8)))  _Float16 v8h;
typedef __attribute__((ext_vector_type(8)))  float    v8f;
typedef __attribute__((ext_vector_type(4)))  unsigned int v4u;
typedef __attribute__((ext_vector_type(4)))  int      v4i;
typedef __attribute__((ext_vector_type(8)))  int      v8i;

constexpr int Bn  = 4;
constexpr int Sn  = 4096;
constexpr int DIN = 768;
constexpr int Dn  = 64;

// kernel-2 LDS tiling
constexpr int KB    = 64;              // keys staged per tile
constexpr int STR   = 72;              // padded row stride (halves), 144B = 9x16B
constexpr int TILEH = KB * STR;        // halves per K or V tile (4608)
constexpr unsigned KOFF0 = 0;                  // buf: K tile [64 keys][STR]
constexpr unsigned VOFF0 = TILEH;              // buf: V^T tile [64 d][STR]
constexpr unsigned BUFH  = 2 * TILEH;          // halves per double-buffer slot
constexpr unsigned POFF  = 2 * BUFH;           // P slabs (4 waves x 16 x 32)

__device__ inline v8f wmma16(v16h a, v16h b, v8f c) {
  // D = A(16x32,f16) * B(32x16,f16) + C(16x16,f32)
  return __builtin_amdgcn_wmma_f32_16x16x32_f16(false, a, false, b, (short)0, c, false, false);
}

// 16-bit A-fragment (16x32) / B-fragment (32x16) loader from a row-major f16
// matrix where lane (lane&15) owns row/column `row`.
// lane<16  : elements [0..7] = K 0..7,  [8..15] = K 16..23
// lane>=16 : elements [0..7] = K 8..15, [8..15] = K 24..31
__device__ inline v16h load_rowfrag_f16(const _Float16* base, int stride,
                                        int row, int k0, int h) {
  const _Float16* p = base + (size_t)row * stride + k0 + h * 8;
  v8h lo = *(const v8h*)p;
  v8h hi = *(const v8h*)(p + 16);
  v16h r;
#pragma unroll
  for (int i = 0; i < 8; ++i) { r[i] = lo[i]; r[8 + i] = hi[i]; }
  return r;
}

// Same fragment pattern but sourced from row-major f32 (converts to f16).
__device__ inline v16h load_rowfrag_f32(const float* base, int stride,
                                        int row, int k0, int h) {
  const float* p = base + (size_t)row * stride + k0 + h * 8;
  v16h r;
#pragma unroll
  for (int i = 0; i < 8; ++i) {
    r[i]     = (_Float16)p[i];
    r[8 + i] = (_Float16)p[16 + i];
  }
  return r;
}

// ---- TDM: 2D tensor_load_to_lds with LDS row padding -> [rows][STR] tile ----
// data_size=2B; tile = 64x64 halves; pad 4 DWORDs after every 32 DWORDs
// (one 128B row) to produce the 144B padded LDS row stride.
__device__ inline void tdm_load_tile(unsigned lds_byte_addr, const void* gaddr,
                                     unsigned tensor_d0, unsigned tensor_d1,
                                     unsigned stride0) {
  const unsigned long long ga = (unsigned long long)gaddr;
  v4u g0;
  g0[0] = 1u;                                             // count=1 (valid user D#)
  g0[1] = lds_byte_addr;                                  // lds_addr
  g0[2] = (unsigned)(ga & 0xFFFFFFFFu);                   // global_addr[31:0]
  g0[3] = (unsigned)((ga >> 32) & 0x1FFFFFFu) | (2u << 30); // addr[56:32] | type=2
  v8i g1;
  // data_size=1(2B) | pad_enable | pad_interval=4(32 DW) | pad_amount=3(4 DW)
  g1[0] = (int)((1u << 16) | (1u << 20) | (4u << 22) | (3u << 25));
  g1[1] = (int)(tensor_d0 << 16);                         // tensor_dim0[15:0] @ bit48
  g1[2] = (int)((tensor_d0 >> 16) | (tensor_d1 << 16));   // dim0 hi / dim1 lo
  g1[3] = (int)((tensor_d1 >> 16) | (64u << 16));         // dim1 hi / tile_dim0=64
  g1[4] = (int)64u;                                       // tile_dim1=64, tile_dim2=0
  g1[5] = (int)stride0;                                   // tensor_dim0_stride[31:0]
  g1[6] = 0;                                              // stride hi / dim1_stride lo
  g1[7] = 0;
  const v4i z4 = {0, 0, 0, 0};
#if __clang_major__ >= 23
  const v8i z8 = {0, 0, 0, 0, 0, 0, 0, 0};
  __builtin_amdgcn_tensor_load_to_lds(g0, g1, z4, z4, z8, 0);
#else
  __builtin_amdgcn_tensor_load_to_lds(g0, g1, z4, z4, 0);
#endif
}

// Stage a 64(d) x 64(keys) slab of V^T (global row stride Sn halves) into a
// padded [64][STR] LDS tile using async global->LDS B128 copies (ASYNCcnt).
// 512 16-byte chunks over 128 threads -> 4 async issues per thread.
__device__ inline void stage_vt_async(const _Float16* __restrict__ vtb, int key0,
                                      unsigned lds_base_bytes, int tid) {
#pragma unroll
  for (int c = 0; c < 4; ++c) {
    const int g  = tid + c * 128;          // chunk id 0..511
    const int d  = g >> 3;                 // V^T row (head dim)
    const int cc = g & 7;                  // 16B chunk within row
    const unsigned lds_off = lds_base_bytes + (unsigned)(d * STR + cc * 8) * 2u;
    const void* gp = (const void*)(vtb + (size_t)d * Sn + key0 + cc * 8);
    asm volatile("global_load_async_to_lds_b128 %0, %1, off"
                 :: "v"(lds_off), "v"(gp) : "memory");
  }
}

__device__ inline void wait_async0() {
  asm volatile("s_wait_asynccnt 0x0" ::: "memory");
}
__device__ inline void wait_tensor0() {
  __builtin_amdgcn_s_wait_tensorcnt(0);
}

// ---------------------------------------------------------------------------
// Kernel 0: convert + transpose weights to f16: wT[w][n][k], n-major rows so
// projection B-fragments become two contiguous 16B loads.
// ---------------------------------------------------------------------------
__global__ void wt_convert_kernel(const float* __restrict__ wq,
                                  const float* __restrict__ wk,
                                  const float* __restrict__ wv,
                                  _Float16* __restrict__ wt) {
  const int idx = blockIdx.x * 256 + threadIdx.x;
  if (idx >= 3 * Dn * DIN) return;
  const int w   = idx / (Dn * DIN);
  const int rem = idx % (Dn * DIN);
  const int n   = rem / DIN;
  const int k   = rem % DIN;
  const float* src = (w == 0) ? wq : (w == 1) ? wk : wv;
  wt[idx] = (_Float16)src[(size_t)k * Dn + n];
}

// ---------------------------------------------------------------------------
// Kernel 1: QKV projection. grid = (B*S)/16 blocks, 96 threads (3 waves).
// Wave 0 -> Q (pre-scaled by 0.125, row-major), wave 1 -> K (row-major),
// wave 2 -> V stored TRANSPOSED as vT[b][d][s] so attention B-fragments are
// contiguous.
// ---------------------------------------------------------------------------
__global__ void qkv_proj_kernel(const float* __restrict__ x,
                                const _Float16* __restrict__ wt,
                                _Float16* __restrict__ qh,
                                _Float16* __restrict__ kh,
                                _Float16* __restrict__ vt) {
  const int wave = threadIdx.x >> 5;
  const int lane = threadIdx.x & 31;
  const int h    = lane >> 4;
  const int ln   = lane & 15;
  const int rowbase = blockIdx.x * 16;

  const _Float16* wtw = wt + (size_t)wave * Dn * DIN;   // [64][768] f16
  v8f acc[4];
#pragma unroll
  for (int t = 0; t < 4; ++t)
#pragma unroll
    for (int j = 0; j < 8; ++j) acc[t][j] = 0.0f;

  for (int k0 = 0; k0 < DIN; k0 += 32) {
    v16h a = load_rowfrag_f32(x, DIN, rowbase + ln, k0, h);
#pragma unroll
    for (int t = 0; t < 4; ++t) {
      v16h b = load_rowfrag_f16(wtw, DIN, t * 16 + ln, k0, h);  // column t*16+ln
      acc[t] = wmma16(a, b, acc[t]);
    }
  }

  if (wave == 2) {
    // V: store transposed, vt[b][d][s]
#pragma unroll
    for (int t = 0; t < 4; ++t)
#pragma unroll
      for (int j = 0; j < 8; ++j) {
        const int row = rowbase + j + h * 8;       // global token row over B*S
        const int bb  = row >> 12;                 // / Sn
        const int s   = row & (Sn - 1);
        vt[((size_t)bb * Dn + t * 16 + ln) * Sn + s] = (_Float16)acc[t][j];
      }
  } else {
    _Float16* dst  = (wave == 0) ? qh : kh;
    const float sc = (wave == 0) ? 0.125f : 1.0f;
#pragma unroll
    for (int t = 0; t < 4; ++t)
#pragma unroll
      for (int j = 0; j < 8; ++j) {
        const int row = rowbase + j + h * 8;       // C layout: VGPR j -> M=j / j+8
        dst[(size_t)row * Dn + t * 16 + ln] = (_Float16)(acc[t][j] * sc);
      }
  }
}

// ---------------------------------------------------------------------------
// Kernel 2: fused flash attention. grid = (S/64, B), 128 threads (4 waves).
// K tiles staged by the Tensor Data Mover (wave 0, TENSORcnt); V^T tiles by
// async global->LDS B128 copies (all threads, ASYNCcnt); double-buffered.
// Each wave owns a 16-query tile with online softmax.
// ---------------------------------------------------------------------------
__global__ void flash_attn_kernel(const _Float16* __restrict__ qh,
                                  const _Float16* __restrict__ kh,
                                  const _Float16* __restrict__ vt,
                                  float* __restrict__ out) {
  __shared__ __align__(16) _Float16 smem[2 * BUFH + 4 * 16 * 32];
  const unsigned smem_base = (unsigned)(size_t)(void*)&smem[0]; // LDS addr = flat[31:0]

  const int tid  = threadIdx.x;
  const int wave = tid >> 5;
  const int lane = tid & 31;
  const int h    = lane >> 4;
  const int ln   = lane & 15;
  const int b    = blockIdx.y;
  const int qbase = blockIdx.x * 64 + wave * 16;

  const _Float16* q   = qh + (size_t)b * Sn * Dn;
  const _Float16* kg  = kh + (size_t)b * Sn * Dn;
  const _Float16* vtb = vt + (size_t)b * Dn * Sn;       // [64][Sn]
  _Float16* Pl = smem + POFF + wave * (16 * 32);        // per-wave 16x32 P slab

  // Q fragments for the two d-halves (K-dim = 64 -> two 16x16x32 steps).
  const v16h aq0 = load_rowfrag_f16(q, Dn, qbase + ln, 0,  h);
  const v16h aq1 = load_rowfrag_f16(q, Dn, qbase + ln, 32, h);

  v8f o[4];
  float m[8], l[8];
#pragma unroll
  for (int j = 0; j < 8; ++j) { m[j] = -INFINITY; l[j] = 0.0f; }
#pragma unroll
  for (int t = 0; t < 4; ++t)
#pragma unroll
    for (int j = 0; j < 8; ++j) o[t][j] = 0.0f;

  // stage first tile into buffer 0: K via TDM, V^T via async copies
  if (wave == 0)
    tdm_load_tile(smem_base + KOFF0 * 2u, kg, Dn, Sn, Dn);
  stage_vt_async(vtb, 0, smem_base + VOFF0 * 2u, tid);

  constexpr int NT = Sn / KB;   // 64 key tiles
  for (int it = 0; it < NT; ++it) {
    wait_async0();                      // my V staging of buf[it&1] done
    if (wave == 0) wait_tensor0();      // K DMA into buf[it&1] done
    __syncthreads();                    // publish; prior reads of other buf done

    if (it + 1 < NT) {                  // prefetch next tile into other buffer
      const unsigned bb = ((unsigned)(it + 1) & 1u) * BUFH;
      if (wave == 0)
        tdm_load_tile(smem_base + (bb + KOFF0) * 2u,
                      kg + (size_t)(it + 1) * KB * Dn, Dn, Sn, Dn);
      stage_vt_async(vtb, (it + 1) * KB, smem_base + (bb + VOFF0) * 2u, tid);
    }

    const _Float16* Kl = smem + ((unsigned)it & 1u) * BUFH + KOFF0; // [64 keys][STR]
    const _Float16* Vl = smem + ((unsigned)it & 1u) * BUFH + VOFF0; // [64 d][STR]

#pragma unroll
    for (int kk = 0; kk < KB; kk += 32) {
      // ---- scores: two 16x16 blocks (keys kk..+15, kk+16..+31 of tile) ----
      v8f s0, s1;
#pragma unroll
      for (int j = 0; j < 8; ++j) { s0[j] = 0.0f; s1[j] = 0.0f; }
      {
        v16h b00 = load_rowfrag_f16(Kl, STR, kk + ln,      0,  h);
        v16h b01 = load_rowfrag_f16(Kl, STR, kk + ln,      32, h);
        v16h b10 = load_rowfrag_f16(Kl, STR, kk + 16 + ln, 0,  h);
        v16h b11 = load_rowfrag_f16(Kl, STR, kk + 16 + ln, 32, h);
        s0 = wmma16(aq0, b00, s0);
        s0 = wmma16(aq1, b01, s0);
        s1 = wmma16(aq0, b10, s1);
        s1 = wmma16(aq1, b11, s1);
      }

      // ---- online softmax over this 32-key block ----
#pragma unroll
      for (int j = 0; j < 8; ++j) {
        float t = fmaxf(s0[j], s1[j]);
        t = fmaxf(t, __shfl_xor(t, 1));
        t = fmaxf(t, __shfl_xor(t, 2));
        t = fmaxf(t, __shfl_xor(t, 4));
        t = fmaxf(t, __shfl_xor(t, 8));        // row max within 16-lane half
        const float mn    = fmaxf(m[j], t);
        const float alpha = __expf(m[j] - mn);
        m[j] = mn;
        const float p0 = __expf(s0[j] - mn);
        const float p1 = __expf(s1[j] - mn);
        float rs = p0 + p1;
        rs += __shfl_xor(rs, 1);
        rs += __shfl_xor(rs, 2);
        rs += __shfl_xor(rs, 4);
        rs += __shfl_xor(rs, 8);
        l[j] = l[j] * alpha + rs;
#pragma unroll
        for (int t4 = 0; t4 < 4; ++t4) o[t4][j] *= alpha;
        // stash P (C layout: VGPR j -> row j+8h, col = ln / 16+ln)
        Pl[(j + h * 8) * 32 + ln]      = (_Float16)p0;
        Pl[(j + h * 8) * 32 + 16 + ln] = (_Float16)p1;
      }

      // same-wave DS ops are in-order; make the RAW explicit
      asm volatile("s_wait_dscnt 0x0" ::: "memory");

      // reload P as an A-fragment (16x32, f16)
      v8h plo = *(const v8h*)&Pl[ln * 32 + h * 8];
      v8h phi = *(const v8h*)&Pl[ln * 32 + 16 + h * 8];
      v16h pa;
#pragma unroll
      for (int i = 0; i < 8; ++i) { pa[i] = plo[i]; pa[8 + i] = phi[i]; }

      // ---- O += P * V : B-frag column d = row of V^T tile, contiguous k ----
#pragma unroll
      for (int t4 = 0; t4 < 4; ++t4) {
        v16h vb = load_rowfrag_f16(Vl, STR, t4 * 16 + ln, kk, h);
        o[t4] = wmma16(pa, vb, o[t4]);
      }
    }
  }

  // ---- normalize and store ----
#pragma unroll
  for (int t4 = 0; t4 < 4; ++t4)
#pragma unroll
    for (int j = 0; j < 8; ++j) {
      const int row = qbase + j + h * 8;
      out[((size_t)b * Sn + row) * Dn + t4 * 16 + ln] = o[t4][j] / l[j];
    }
}

extern "C" void kernel_launch(void* const* d_in, const int* in_sizes, int n_in,
                              void* d_out, int out_size, void* d_ws, size_t ws_size,
                              hipStream_t stream) {
  const float* x  = (const float*)d_in[0];
  const float* wq = (const float*)d_in[1];
  const float* wk = (const float*)d_in[2];
  const float* wv = (const float*)d_in[3];

  _Float16* qh = (_Float16*)d_ws;                       // [B*S, 64] f16 (pre-scaled)
  _Float16* kh = qh + (size_t)Bn * Sn * Dn;             // [B*S, 64] f16
  _Float16* vt = kh + (size_t)Bn * Sn * Dn;             // [B][64][S] f16 (V^T)
  _Float16* wt = vt + (size_t)Bn * Sn * Dn;             // [3][64][768] f16 transposed

  {
    const int total = 3 * Dn * DIN;
    wt_convert_kernel<<<dim3((total + 255) / 256), dim3(256), 0, stream>>>(
        wq, wk, wv, wt);
  }

  qkv_proj_kernel<<<dim3((Bn * Sn) / 16), dim3(96), 0, stream>>>(
      x, wt, qh, kh, vt);

  flash_attn_kernel<<<dim3(Sn / KB, Bn), dim3(128), 0, stream>>>(
      qh, kh, vt, (float*)d_out);
}
